// DCN_51539607711
// MI455X (gfx1250) — matrix operand order repaired and checked by reading, not compiled, not run on hardware
//
#include <hip/hip_runtime.h>

// Mixed low-rank DCN-v2 (L=3, E=4, D=512, R=64, B=16384).
// gate = softmax over a size-1 axis == 1.0 -> G input is mathematically dead.
// Per layer: xl = x0 * (cv @ W2 + E*bias) + xl, where
//   v  = tanh(xl @ Vcat)                      [B, 256]
//   cv = tanh(blockdiag_e(C_e) applied to v)  [B, 256]
//   W2[e*R+r, d] = U[e, d, r]                 [256, 512]
// bf16 WMMA (v_wmma_f32_16x16x32_bf16) with f32 accumulation; weights are
// pre-converted/pre-transposed to bf16 once per launch; LDS staging uses
// gfx1250 async global->LDS copies (ASYNCcnt) when available.

#define BDIM     16384
#define DD       512
#define RR       64
#define EE       4
#define N1       256
#define TM       32
#define NTHREADS 256   // 8 wave32 waves

// per-layer preconverted bf16 weight block sizes (elements)
#define VT_E 131072                     // [16 kc][256 n][32 kk]
#define CT_E 16384                      // [4 e][64 s][64 r]
#define UT_E 131072                     // [8 (e,h)][512 d][32 rr]
#define LAYER_E (VT_E + CT_E + UT_E)    // 278528

typedef __bf16 bf16_t;
typedef __attribute__((ext_vector_type(16))) __bf16        v16bf;
typedef __attribute__((ext_vector_type(8)))  float         v8f;
typedef __attribute__((ext_vector_type(4)))  unsigned int  u32x4;
typedef __attribute__((ext_vector_type(2)))  unsigned int  u32x2;
typedef __attribute__((ext_vector_type(4)))  int           i32x4;

#if __has_builtin(__builtin_amdgcn_global_load_async_to_lds_b128) && \
    __has_builtin(__builtin_amdgcn_s_wait_asynccnt)
#define HAVE_ASYNC_LDS 1
#else
#define HAVE_ASYNC_LDS 0
#endif

union Frag { v16bf v; u32x4 q[2]; };

__device__ __forceinline__ unsigned short f2bf_raw(float f) {
  unsigned u = __builtin_bit_cast(unsigned, f);
  return (unsigned short)((u + 0x7fffu + ((u >> 16) & 1u)) >> 16);  // RNE
}
__device__ __forceinline__ bf16_t f2bf(float f) {
  unsigned short h = f2bf_raw(f);
  return __builtin_bit_cast(bf16_t, h);
}

// gfx1250 has native V_TANH_F32 (TRANS op). Fallback: branchless exp2/rcp
// form; |error| well below the bf16 quantization applied right after.
__device__ __forceinline__ float fast_tanh(float x) {
#if __has_builtin(__builtin_amdgcn_tanhf)
  return __builtin_amdgcn_tanhf(x);
#else
  float a = __builtin_fabsf(x);
  float t = __builtin_amdgcn_exp2f(-2.8853900817779268f * a);  // e^(-2a)
  float r = (1.0f - t) * __builtin_amdgcn_rcpf(1.0f + t);
  return __builtin_copysignf(r, x);
#endif
}

// 16-byte global->LDS copy: async DMA path (tracked by ASYNCcnt) if present.
// Clang's builtin takes pointers-to-v4i32 in the global (AS1) and LDS (AS3)
// address spaces.
__device__ __forceinline__ void g2l_b128(const bf16_t* g, bf16_t* l) {
#if HAVE_ASYNC_LDS
  __builtin_amdgcn_global_load_async_to_lds_b128(
      (__attribute__((address_space(1))) i32x4*)g,
      (__attribute__((address_space(3))) i32x4*)l, 0, 0);
#else
  *(u32x4*)l = *(const u32x4*)g;
#endif
}
__device__ __forceinline__ void async_fence() {
#if HAVE_ASYNC_LDS
  __builtin_amdgcn_s_wait_asynccnt(0);
#endif
}

// A-operand fragment (16-bit A 16x32): per-lane K = kh*8+[0,8) and +16
__device__ __forceinline__ v16bf ld_fragA(const bf16_t* p) {
  Frag f;
  f.q[0] = *(const u32x4*)(p);
  f.q[1] = *(const u32x4*)(p + 16);
  return f.v;
}
// B-operand fragment (16-bit B 32x16): per-lane K = kh*8+[0,8) and +16
__device__ __forceinline__ v16bf ld_fragB(const bf16_t* p) {
  Frag f;
  f.q[0] = *(const u32x4*)(p);
  f.q[1] = *(const u32x4*)(p + 16);
  return f.v;
}

// ---- one-shot weight prep: fp32 -> bf16, pre-transposed & pre-chunked ----
__global__ __launch_bounds__(NTHREADS)
void prep_weights_kernel(const float* __restrict__ U, const float* __restrict__ V,
                         const float* __restrict__ C, bf16_t* __restrict__ wbuf) {
  int idx = blockIdx.x * NTHREADS + threadIdx.x;
  const int TOT = 3 * LAYER_E;
  if (idx >= TOT) return;
  int l = idx / LAYER_E;
  int r = idx - l * LAYER_E;
  float val;
  if (r < VT_E) {                       // Vtp[kc][n][kk] = V[l,e,kc*32+kk,rr]
    int kc = r >> 13, rem = r & 8191, n = rem >> 5, kk = rem & 31;
    int e = n >> 6, rr = n & 63;
    val = V[((size_t)(l * EE + e) * DD + kc * 32 + kk) * RR + rr];
  } else if (r < VT_E + CT_E) {         // Ctp[e][s][r] = C[l,e,r,s]
    int q = r - VT_E;
    int e = q >> 12, rem = q & 4095, s = rem >> 6, rr = rem & 63;
    val = C[((size_t)(l * EE + e) * RR + rr) * RR + s];
  } else {                              // Utp[e*2+h][d][rr] = U[l,e,d,h*32+rr]
    int q = r - (VT_E + CT_E);
    int eh = q >> 14, rem = q & 16383, d = rem >> 5, rr = rem & 31;
    int e = eh >> 1, h = eh & 1;
    val = U[((size_t)(l * EE + e) * DD + d) * RR + h * 32 + rr];
  }
  wbuf[idx] = f2bf(val);
}

// ---- fused per-layer kernel ----
__global__ __launch_bounds__(NTHREADS)
void dcn_layer_kernel(const float* __restrict__ x0,
                      const float* __restrict__ xin,
                      const bf16_t* __restrict__ Vtp,
                      const bf16_t* __restrict__ Ctp,
                      const bf16_t* __restrict__ Utp,
                      const float* __restrict__ biasl,
                      float* __restrict__ xout) {
  __shared__ __align__(16) bf16_t smem[26624];  // 52 KB arena
  bf16_t* sAc  = smem;           // P1: 2 x [32][32]  X chunks (double buffer)
  bf16_t* sBt  = smem + 2048;    // P1: 2 x [256][32] V chunks (double buffer)
  bf16_t* vbuf = smem + 18432;   // [32][256]: v, then cv in place
  bf16_t* sW   = smem;           // P2: C^T [4][64][64]; P3: U chunk [512][32]

  const int t    = threadIdx.x;
  const int lane = t & 31, w = t >> 5;
  const int lm   = lane & 15, kh = lane >> 4;
  const int row0 = blockIdx.x * TM;

  auto stage1 = [&](int kc, int b) {
    // B chunk: contiguous 8 KB of preconverted bf16 -> async DMA
    const bf16_t* src = Vtp + kc * 8192;
    bf16_t* dst = sBt + b * 8192;
    #pragma unroll
    for (int it = 0; it < 2; ++it) {
      int li = (it * NTHREADS + t) * 8;
      g2l_b128(src + li, dst + li);
    }
    // A chunk: [32 rows][32 k] fp32 -> bf16 (one float4 per thread)
    int rr = t >> 3, c4 = (t & 7) * 4;
    float4 v4 = *(const float4*)(xin + (size_t)(row0 + rr) * DD + kc * 32 + c4);
    u32x2 pk = {(unsigned)f2bf_raw(v4.x) | ((unsigned)f2bf_raw(v4.y) << 16),
                (unsigned)f2bf_raw(v4.z) | ((unsigned)f2bf_raw(v4.w) << 16)};
    *(u32x2*)(sAc + b * 1024 + rr * 32 + c4) = pk;
  };

  // ---------- Phase 1: v_pre = X @ Vcat, [32 x 256], K = 512 ----------
  v8f acc[4];
  #pragma unroll
  for (int i = 0; i < 4; ++i) acc[i] = (v8f){0,0,0,0,0,0,0,0};

  stage1(0, 0);
  for (int kc = 0; kc < 16; ++kc) {
    async_fence();
    __syncthreads();                       // chunk kc ready; buffer kc^1 free
    if (kc + 1 < 16) stage1(kc + 1, (kc + 1) & 1);
    const bf16_t* sa = sAc + (kc & 1) * 1024;
    const bf16_t* sb = sBt + (kc & 1) * 8192;
    #pragma unroll
    for (int i = 0; i < 4; ++i) {
      int tt = w + 8 * i;
      int m0 = (tt >> 4) << 4, n0 = (tt & 15) << 4;
      v16bf a = ld_fragA(sa + (m0 + lm) * 32 + kh * 8);
      v16bf b = ld_fragB(sb + (n0 + lm) * 32 + kh * 8);
      acc[i] = __builtin_amdgcn_wmma_f32_16x16x32_bf16(
                   false, a, false, b, (short)0, acc[i], false, false);
    }
  }
  __syncthreads();  // all waves done with P1 LDS before C^T overwrites it

  // ---------- Phase 2: cv = tanh(blockdiag(C) @ v) ----------
  // stage C^T (32 KB, contiguous) while writing v = tanh(acc) to vbuf
  #pragma unroll
  for (int it = 0; it < 8; ++it) {
    int li = (it * NTHREADS + t) * 8;
    g2l_b128(Ctp + li, sW + li);
  }
  #pragma unroll
  for (int i = 0; i < 4; ++i) {
    int tt = w + 8 * i;
    int m0 = (tt >> 4) << 4, n0 = (tt & 15) << 4;
    #pragma unroll
    for (int j = 0; j < 8; ++j)
      vbuf[(m0 + j + (kh << 3)) * N1 + n0 + lm] = f2bf(fast_tanh(acc[i][j]));
  }
  async_fence();
  __syncthreads();

  v8f acc2[4];
  #pragma unroll
  for (int i = 0; i < 4; ++i) acc2[i] = (v8f){0,0,0,0,0,0,0,0};
  #pragma unroll
  for (int i = 0; i < 4; ++i) {
    int tt = w + 8 * i;
    int m0 = (tt >> 4) << 4, n0 = (tt & 15) << 4;
    int e = n0 >> 6, nl = (n0 & 63) + lm;
    #pragma unroll
    for (int s = 0; s < 2; ++s) {       // K = 64 per expert, two 32-chunks
      v16bf a = ld_fragA(vbuf + (m0 + lm) * N1 + e * 64 + s * 32 + kh * 8);
      v16bf b = ld_fragB(sW + e * 4096 + nl * 64 + s * 32 + kh * 8);
      acc2[i] = __builtin_amdgcn_wmma_f32_16x16x32_bf16(
                    false, a, false, b, (short)0, acc2[i], false, false);
    }
  }
  __syncthreads();  // all v reads done; write cv in place
  #pragma unroll
  for (int i = 0; i < 4; ++i) {
    int tt = w + 8 * i;
    int m0 = (tt >> 4) << 4, n0 = (tt & 15) << 4;
    #pragma unroll
    for (int j = 0; j < 8; ++j)
      vbuf[(m0 + j + (kh << 3)) * N1 + n0 + lm] = f2bf(fast_tanh(acc2[i][j]));
  }
  __syncthreads();

  // ---------- Phase 3: out = cv @ W2, [32 x 512], K = 256 ----------
  v8f acc3[8];
  #pragma unroll
  for (int i = 0; i < 8; ++i) acc3[i] = (v8f){0,0,0,0,0,0,0,0};

  for (int c = 0; c < 8; ++c) {         // (e, r-half) chunks
    // stage U chunk (32 KB, contiguous)
    #pragma unroll
    for (int it = 0; it < 8; ++it) {
      int li = (it * NTHREADS + t) * 8;
      g2l_b128(Utp + c * 16384 + li, sW + li);
    }
    async_fence();
    __syncthreads();
    int e = c >> 1, h = c & 1;
    #pragma unroll
    for (int i = 0; i < 8; ++i) {
      int tt = w + 8 * i;
      int m0 = (tt >> 5) << 4, n0 = (tt & 31) << 4;
      v16bf a = ld_fragA(vbuf + (m0 + lm) * N1 + e * 64 + h * 32 + kh * 8);
      v16bf b = ld_fragB(sW + (n0 + lm) * 32 + kh * 8);
      acc3[i] = __builtin_amdgcn_wmma_f32_16x16x32_bf16(
                    false, a, false, b, (short)0, acc3[i], false, false);
    }
    __syncthreads();
  }

  // ---------- epilogue: xout = x0 * (acc + E*bias) + xin ----------
  #pragma unroll
  for (int i = 0; i < 8; ++i) {
    int tt = w + 8 * i;
    int m0 = (tt >> 5) << 4, n0 = (tt & 31) << 4;
    int n = n0 + lm;
    float bsum = (float)EE * biasl[n];
    #pragma unroll
    for (int j = 0; j < 8; ++j) {
      int m = m0 + j + (kh << 3);
      size_t idx = (size_t)(row0 + m) * DD + n;
      xout[idx] = x0[idx] * (acc3[i][j] + bsum) + xin[idx];
    }
  }
}

extern "C" void kernel_launch(void* const* d_in, const int* in_sizes, int n_in,
                              void* d_out, int out_size, void* d_ws, size_t ws_size,
                              hipStream_t stream) {
  (void)in_sizes; (void)n_in; (void)out_size; (void)ws_size;
  const float* x    = (const float*)d_in[0];
  const float* U    = (const float*)d_in[1];
  const float* V    = (const float*)d_in[2];
  const float* C    = (const float*)d_in[3];
  // d_in[4] (G) is mathematically dead: softmax over a size-1 axis == 1.0
  const float* bias = (const float*)d_in[5];
  float* out = (float*)d_out;

  bf16_t* wbuf = (bf16_t*)d_ws;  // 3 * 278528 * 2 B ~= 1.6 MB of scratch

  const int TOT = 3 * LAYER_E;
  prep_weights_kernel<<<(TOT + NTHREADS - 1) / NTHREADS, NTHREADS, 0, stream>>>(
      U, V, C, wbuf);

  dim3 grid(BDIM / TM), block(NTHREADS);
  // Blocks own disjoint 32-row stripes and read them before writing them,
  // so the xl update is safe in-place: layer0 x->out, layers 1,2 in-place.
  for (int l = 0; l < 3; ++l) {
    const bf16_t* wl = wbuf + (size_t)l * LAYER_E;
    dcn_layer_kernel<<<grid, block, 0, stream>>>(
        x, (l == 0) ? x : out, wl, wl + VT_E, wl + VT_E + CT_E,
        bias + (size_t)l * DD, out);
  }
}
